// GATEncoder_81544249081904
// MI455X (gfx1250) — matrix operand order repaired and checked by reading, not compiled, not run on hardware
//
#include <hip/hip_runtime.h>
#include <hip/hip_bf16.h>
#include <math.h>

// ---------------------------------------------------------------------------
// GAT 2-layer encoder for MI455X (gfx1250, wave32).
// GEMMs: v_wmma_f32_16x16x32_bf16, LDS-staged bf16 weights (ds_load_b128),
// fully unrolled K loop. Edge softmax-aggregate: f32 global atomics.
// ---------------------------------------------------------------------------

#define N_NODES   50000
#define N_EDGES   800000
#define IN_C      128
#define HID       64
#define HEADS     4
#define OUT_C     32
#define NEG_SLOPE 0.2f
#define EPS_F     1e-16f

typedef __attribute__((ext_vector_type(16))) __bf16 v16bf;
typedef __attribute__((ext_vector_type(8)))  __bf16 v8bf;
typedef __attribute__((ext_vector_type(8)))  float  v8f;

static inline int ceil_div(long long a, long long b) { return (int)((a + b - 1) / b); }

// ---------------------------------------------------------------------------
// WMMA GEMM: D[M,N] = A[M,K] * B[K,N], row-major f32 in memory, bf16 compute,
// f32 accumulate. K, N compile-time; M % 16 == 0 (50000 = 3125*16).
//
// Each 256-thread block (8 waves) stages the FULL weight matrix B into LDS
// transposed + converted to bf16 (Bt[n][k]), so each wave's B fragment is a
// pair of 16-byte LDS runs (2x ds_load_b128). Each wave then computes one
// 16x16 D tile with K/32 back-to-back WMMAs.
// ---------------------------------------------------------------------------
template <int K, int N>
__global__ void gemm_wmma_bf16(const float* __restrict__ A,
                               const float* __restrict__ B,
                               float* __restrict__ D,
                               int M) {
    extern __shared__ __bf16 ldsB[];   // Bt[N][K], bf16: N*K*2 bytes

    // ---- cooperative stage: Bt[n*K + k] = bf16(B[k*N + n]) ----
    for (int i = threadIdx.x; i < K * N; i += blockDim.x) {
        const int k = i / N;
        const int n = i % N;
        ldsB[n * K + k] = (__bf16)B[(size_t)k * N + n];
    }
    __syncthreads();   // s_barrier_signal/wait (workgroup barrier)

    const int lane   = threadIdx.x & 31;
    const int wid    = blockIdx.x * 8 + (threadIdx.x >> 5);
    const int ntiles = N >> 4;
    const int mtile  = wid / ntiles;
    const int ntile  = wid % ntiles;
    if (mtile * 16 >= M) return;   // after barrier: safe

    const int m  = lane & 15;   // A row within tile
    const int n  = lane & 15;   // B/D col within tile
    const int kg = lane >> 4;   // lane-half selector

    const float* Arow = A + (size_t)(mtile * 16 + m) * K;
    const __bf16* Brow = ldsB + (size_t)(ntile * 16 + n) * K;

    v8f acc = {};
#pragma unroll
    for (int k0 = 0; k0 < K; k0 += 32) {
        // --- A fragment: 16-bit A-matrix 16x32 layout (ISA 7.12.2).
        // Lane needs K = kg*8+[0..7] and 16+kg*8+[0..7]: two 8-float runs
        // -> four global_load_b128.
        const float4* A4 = (const float4*)(Arow + k0);
        const float4 f0 = A4[kg * 2];
        const float4 f1 = A4[kg * 2 + 1];
        const float4 f2 = A4[4 + kg * 2];
        const float4 f3 = A4[5 + kg * 2];
        v16bf a;
        a[0]  = (__bf16)f0.x;  a[1]  = (__bf16)f0.y;
        a[2]  = (__bf16)f0.z;  a[3]  = (__bf16)f0.w;
        a[4]  = (__bf16)f1.x;  a[5]  = (__bf16)f1.y;
        a[6]  = (__bf16)f1.z;  a[7]  = (__bf16)f1.w;
        a[8]  = (__bf16)f2.x;  a[9]  = (__bf16)f2.y;
        a[10] = (__bf16)f2.z;  a[11] = (__bf16)f2.w;
        a[12] = (__bf16)f3.x;  a[13] = (__bf16)f3.y;
        a[14] = (__bf16)f3.z;  a[15] = (__bf16)f3.w;

        // --- B fragment: lane-half kg needs K = kg*8+[0..7] and 16+kg*8+[0..7]
        // at column n: two 16-byte runs in Bt -> 2x ds_load_b128.
        const v8bf* bp = (const v8bf*)(Brow + k0 + kg * 8);
        const v8bf blo = bp[0];
        const v8bf bhi = bp[2];
        v16bf b;
#pragma unroll
        for (int i = 0; i < 8; ++i) { b[i] = blo[i]; b[i + 8] = bhi[i]; }

        acc = __builtin_amdgcn_wmma_f32_16x16x32_bf16(
            /*neg_a=*/false, a, /*neg_b=*/false, b,
            /*c_mod=*/(short)0, acc, /*reuse_a=*/false, /*reuse_b=*/false);
    }

    // --- D tile: 32-bit C/D layout: VGPR v -> row v + 8*kg, col n ---
    float* Drow = D + (size_t)(mtile * 16) * N + ntile * 16 + n;
#pragma unroll
    for (int v = 0; v < 8; ++v) {
        Drow[(size_t)(v + kg * 8) * N] = acc[v];
    }
}

// ---------------------------------------------------------------------------
// a_src[n,h] = sum_c h[n,h,c]*att_src[h,c]; same for a_dst. 1 thread / (n,h).
// ---------------------------------------------------------------------------
__global__ void node_alpha_kernel(const float* __restrict__ h,
                                  const float* __restrict__ att_src,
                                  const float* __restrict__ att_dst,
                                  float* __restrict__ a_src,
                                  float* __restrict__ a_dst,
                                  int nNodes, int nHeads, int C) {
    const int idx = blockIdx.x * blockDim.x + threadIdx.x;
    if (idx >= nNodes * nHeads) return;
    const int hd = idx % nHeads;
    const float* row = h + (size_t)idx * C;
    const float* ws  = att_src + (size_t)hd * C;
    const float* wd  = att_dst + (size_t)hd * C;
    float ss = 0.f, sd = 0.f;
    for (int c = 0; c < C; ++c) {
        const float v = row[c];
        ss += v * ws[c];
        sd += v * wd[c];
    }
    a_src[idx] = ss;
    a_dst[idx] = sd;
}

__global__ void fill_kernel(float* __restrict__ p, float val, long long n) {
    const long long i = blockIdx.x * (long long)blockDim.x + threadIdx.x;
    if (i < n) p[i] = val;
}

__device__ inline void atomicMaxF(float* address, float val) {
    unsigned int* ua = (unsigned int*)address;
    unsigned int old = *ua, assumed;
    do {
        assumed = old;
        if (__uint_as_float(assumed) >= val) break;
        old = atomicCAS(ua, assumed, __float_as_uint(val));
    } while (old != assumed);
}

// Pass 1: alpha = leaky_relu(a_src[src]+a_dst[dst]); amax[dst] = max(...).
// eTotal = E + N (self loops appended). 1 thread per (edge, head).
__global__ void edge_alpha_max(const int* __restrict__ srcArr,
                               const int* __restrict__ dstArr,
                               const float* __restrict__ a_src,
                               const float* __restrict__ a_dst,
                               float* __restrict__ alpha,
                               float* __restrict__ amax,
                               int eTotal, int eOrig, int nHeads) {
    const long long idx = blockIdx.x * (long long)blockDim.x + threadIdx.x;
    if (idx >= (long long)eTotal * nHeads) return;
    const int e  = (int)(idx / nHeads);
    const int hd = (int)(idx % nHeads);
    const int s  = (e < eOrig) ? srcArr[e] : (e - eOrig);
    const int d  = (e < eOrig) ? dstArr[e] : (e - eOrig);
    float al = a_src[(size_t)s * nHeads + hd] + a_dst[(size_t)d * nHeads + hd];
    al = (al >= 0.f) ? al : NEG_SLOPE * al;
    alpha[idx] = al;
    atomicMaxF(&amax[(size_t)d * nHeads + hd], al);
}

// Pass 2: ex = exp(alpha - amax[dst]); denom[dst] += ex.
__global__ void edge_exp_sum(const int* __restrict__ dstArr,
                             float* __restrict__ alpha,
                             const float* __restrict__ amax,
                             float* __restrict__ denom,
                             int eTotal, int eOrig, int nHeads) {
    const long long idx = blockIdx.x * (long long)blockDim.x + threadIdx.x;
    if (idx >= (long long)eTotal * nHeads) return;
    const int e  = (int)(idx / nHeads);
    const int hd = (int)(idx % nHeads);
    const int d  = (e < eOrig) ? dstArr[e] : (e - eOrig);
    const float ex = __expf(alpha[idx] - amax[(size_t)d * nHeads + hd]);
    alpha[idx] = ex;
    atomicAdd(&denom[(size_t)d * nHeads + hd], ex);
}

// Pass 3: out[dst, h, c] += h[src, h, c] * ex/(denom[dst,h]+eps).
// One wave32 per edge; lanes stride the F = nHeads*C features.
__global__ void edge_aggregate(const int* __restrict__ srcArr,
                               const int* __restrict__ dstArr,
                               const float* __restrict__ hFeat,
                               const float* __restrict__ alpha,  // holds ex
                               const float* __restrict__ denom,
                               float* __restrict__ out,
                               int eTotal, int eOrig, int nHeads, int C) {
    const int lane = threadIdx.x & 31;
    const long long wid = (blockIdx.x * (long long)blockDim.x + threadIdx.x) >> 5;
    if (wid >= eTotal) return;
    const int e = (int)wid;
    const int s = (e < eOrig) ? srcArr[e] : (e - eOrig);
    const int d = (e < eOrig) ? dstArr[e] : (e - eOrig);
    const int F = nHeads * C;
    const float* hrow = hFeat + (size_t)s * F;
    float* orow = out + (size_t)d * F;
    __builtin_prefetch(hrow, 0, 1);   // global_prefetch_b8
    for (int f = lane; f < F; f += 32) {
        const int hd = f / C;
        const float coef = alpha[(size_t)e * nHeads + hd] /
                           (denom[(size_t)d * nHeads + hd] + EPS_F);
        atomicAdd(&orow[f], hrow[f] * coef);
    }
}

// Layer-1 epilogue: x = x + bias; x = elu(x) (alpha=1), in place.
__global__ void bias_elu_kernel(float* __restrict__ x,
                                const float* __restrict__ bias,
                                long long total, int F) {
    const long long i = blockIdx.x * (long long)blockDim.x + threadIdx.x;
    if (i >= total) return;
    const float v = x[i] + bias[i % F];
    x[i] = (v > 0.f) ? v : expm1f(v);
}

// Layer-2 epilogue: out += bias (heads==1 so mean over heads is identity).
__global__ void bias_add_kernel(float* __restrict__ x,
                                const float* __restrict__ bias,
                                long long total, int F) {
    const long long i = blockIdx.x * (long long)blockDim.x + threadIdx.x;
    if (i >= total) return;
    x[i] += bias[i % F];
}

// ---------------------------------------------------------------------------
extern "C" void kernel_launch(void* const* d_in, const int* in_sizes, int n_in,
                              void* d_out, int out_size, void* d_ws, size_t ws_size,
                              hipStream_t stream) {
    const float* x        = (const float*)d_in[0];
    const int*   ei       = (const int*)  d_in[1];   // [2, E] flat: src then dst
    const float* W1       = (const float*)d_in[2];   // [128, 256]
    const float* att_src1 = (const float*)d_in[3];   // [4, 64]
    const float* att_dst1 = (const float*)d_in[4];
    const float* b1       = (const float*)d_in[5];   // [256]
    const float* W2       = (const float*)d_in[6];   // [256, 32]
    const float* att_src2 = (const float*)d_in[7];   // [1, 32]
    const float* att_dst2 = (const float*)d_in[8];
    const float* b2       = (const float*)d_in[9];   // [32]
    float*       out      = (float*)d_out;           // [N, 32]

    const int* srcArr = ei;
    const int* dstArr = ei + N_EDGES;

    const int N  = N_NODES;
    const int E  = N_EDGES;
    const int ET = E + N;            // with self loops
    const int F1 = HEADS * HID;      // 256

    // ---- workspace layout (floats) ----
    float* ws    = (float*)d_ws;
    float* h1    = ws;                          // N*256
    float* out1  = h1   + (size_t)N * F1;       // N*256 (aggregate -> elu'd)
    float* aSrc  = out1 + (size_t)N * F1;       // N*HEADS (reused layer 2)
    float* aDst  = aSrc + (size_t)N * HEADS;
    float* amax  = aDst + (size_t)N * HEADS;
    float* denom = amax + (size_t)N * HEADS;
    float* alpha = denom + (size_t)N * HEADS;   // ET*HEADS (reused layer 2)
    float* h2    = alpha + (size_t)ET * HEADS;  // N*32
    (void)ws_size; (void)n_in; (void)in_sizes; (void)out_size;

    const int TB = 256;

    // ================= Layer 1 (HEADS=4, C=HID=64) =================
    {
        const long long tiles = (long long)(N / 16) * (F1 / 16);  // 50000
        gemm_wmma_bf16<IN_C, HEADS * HID>
            <<<ceil_div(tiles, 8), TB, IN_C * F1 * sizeof(__bf16), stream>>>(
                x, W1, h1, N);

        node_alpha_kernel<<<ceil_div((long long)N * HEADS, TB), TB, 0, stream>>>(
            h1, att_src1, att_dst1, aSrc, aDst, N, HEADS, HID);

        fill_kernel<<<ceil_div((long long)N * HEADS, TB), TB, 0, stream>>>(
            amax, -INFINITY, (long long)N * HEADS);
        fill_kernel<<<ceil_div((long long)N * HEADS, TB), TB, 0, stream>>>(
            denom, 0.f, (long long)N * HEADS);
        fill_kernel<<<ceil_div((long long)N * F1, TB), TB, 0, stream>>>(
            out1, 0.f, (long long)N * F1);

        edge_alpha_max<<<ceil_div((long long)ET * HEADS, TB), TB, 0, stream>>>(
            srcArr, dstArr, aSrc, aDst, alpha, amax, ET, E, HEADS);
        edge_exp_sum<<<ceil_div((long long)ET * HEADS, TB), TB, 0, stream>>>(
            dstArr, alpha, amax, denom, ET, E, HEADS);
        edge_aggregate<<<ceil_div((long long)ET * 32, TB), TB, 0, stream>>>(
            srcArr, dstArr, h1, alpha, denom, out1, ET, E, HEADS, HID);

        bias_elu_kernel<<<ceil_div((long long)N * F1, TB), TB, 0, stream>>>(
            out1, b1, (long long)N * F1, F1);
    }

    // ================= Layer 2 (HEADS=1, C=OUT_C=32) =================
    {
        const long long tiles = (long long)(N / 16) * (OUT_C / 16);  // 6250
        gemm_wmma_bf16<HEADS * HID, OUT_C>
            <<<ceil_div(tiles, 8), TB, F1 * OUT_C * sizeof(__bf16), stream>>>(
                out1, W2, h2, N);

        node_alpha_kernel<<<ceil_div((long long)N, TB), TB, 0, stream>>>(
            h2, att_src2, att_dst2, aSrc, aDst, N, 1, OUT_C);

        fill_kernel<<<ceil_div((long long)N, TB), TB, 0, stream>>>(
            amax, -INFINITY, (long long)N);
        fill_kernel<<<ceil_div((long long)N, TB), TB, 0, stream>>>(
            denom, 0.f, (long long)N);
        fill_kernel<<<ceil_div((long long)N * OUT_C, TB), TB, 0, stream>>>(
            out, 0.f, (long long)N * OUT_C);

        edge_alpha_max<<<ceil_div((long long)ET, TB), TB, 0, stream>>>(
            srcArr, dstArr, aSrc, aDst, alpha, amax, ET, E, 1);
        edge_exp_sum<<<ceil_div((long long)ET, TB), TB, 0, stream>>>(
            dstArr, alpha, amax, denom, ET, E, 1);
        edge_aggregate<<<ceil_div((long long)ET * 32, TB), TB, 0, stream>>>(
            srcArr, dstArr, h2, alpha, denom, out, ET, E, 1, OUT_C);

        bias_add_kernel<<<ceil_div((long long)N * OUT_C, TB), TB, 0, stream>>>(
            out, b2, (long long)N * OUT_C, OUT_C);
    }
}
